// SimpleAttention_60258391163521
// MI455X (gfx1250) — matrix operand order repaired and checked by reading, not compiled, not run on hardware
//
#include <hip/hip_runtime.h>
#include <hip/hip_bf16.h>

// MLA absorbed-attention decode, CDNA5 (gfx1250), wave32.
// K1: q-proj + k-absorb (WMMA bf16).  K2: fused flash attention, KV tile DMA'd
// into LDS by the Tensor Data Mover (tensor_load_to_lds / s_wait_tensorcnt),
// scores+context via WMMA.  K3: v-absorb output projection (WMMA).
// Workspace: [0,8MB) q_absorbed f32 [B][H][KR]; [8MB,16MB) context f32 [B][H][KR].

#define B_   32
#define H_   128
#define D_   128
#define QR_  1536
#define KR_  512
#define S_   4096

typedef __attribute__((ext_vector_type(16))) __bf16       v16bf;
typedef __attribute__((ext_vector_type(8)))  float        v8f;
typedef __attribute__((ext_vector_type(4)))  unsigned int v4u;
typedef __attribute__((ext_vector_type(8)))  int          v8i;
typedef __attribute__((ext_vector_type(4)))  int          v4i;

union BF8 { uint4 u; __bf16 h[8]; };
union BF4 { uint2 u; __bf16 h[4]; };

// ---------------------------------------------------------------------------
// WMMA wrapper
// ---------------------------------------------------------------------------
__device__ __forceinline__ v8f wmma_bf16(v16bf a, v16bf b, v8f c) {
  return __builtin_amdgcn_wmma_f32_16x16x32_bf16(
      false, a, false, b, (short)0, c, false, false);
}

// ---------------------------------------------------------------------------
// Fragment builders (wave32; contiguous k order, shared by A and B)
// A 16x32 MxK: lane = m (0..15), lane[4] picks K-half; VGPR v holds
//   k = (lane>>4)*16 + 2v .. +1                      -> one contiguous 16-elt run.
// B 32x16 KxN: lane = n, lane[4] picks K-half; VGPR v holds
//   k = (lane>>4)*16 + 2v .. +1                      -> one contiguous 16-elt run.
// ---------------------------------------------------------------------------

// A from f32 global, row-major, contiguous in k: two pairs of float4 loads.
__device__ __forceinline__ v16bf load_a_f32v(const float* __restrict__ base,
                                             long ldm, int m0, int k0, int lane) {
  const int m  = m0 + (lane & 15);
  const int kh = lane >> 4;
  const float4* p = (const float4*)(base + (long)m * ldm + k0 + kh * 16);
  const float4 x0 = p[0], x1 = p[1];          // k rel 0..7
  const float4 y0 = p[2], y1 = p[3];          // k rel 8..15 (= +8 floats)
  v16bf a;
  a[0]=(__bf16)x0.x; a[1]=(__bf16)x0.y; a[2]=(__bf16)x0.z; a[3]=(__bf16)x0.w;
  a[4]=(__bf16)x1.x; a[5]=(__bf16)x1.y; a[6]=(__bf16)x1.z; a[7]=(__bf16)x1.w;
  a[8]=(__bf16)y0.x; a[9]=(__bf16)y0.y; a[10]=(__bf16)y0.z; a[11]=(__bf16)y0.w;
  a[12]=(__bf16)y1.x; a[13]=(__bf16)y1.y; a[14]=(__bf16)y1.z; a[15]=(__bf16)y1.w;
  return a;
}

// A from bf16 LDS, contiguous in k: two b128 loads.
__device__ __forceinline__ v16bf load_a_bf16v(const __bf16* __restrict__ base,
                                              int ldm, int m0, int k0, int lane) {
  const int m  = m0 + (lane & 15);
  const int kh = lane >> 4;
  const __bf16* p = base + m * ldm + k0 + kh * 16;
  BF8 lo, hi;
  lo.u = *(const uint4*)p;
  hi.u = *(const uint4*)(p + 8);
  v16bf a;
#pragma unroll
  for (int i = 0; i < 8; ++i) { a[i] = lo.h[i]; a[8 + i] = hi.h[i]; }
  return a;
}

// B from f32 memory with unit k-stride (element (k,n) = base[n*sn + k]): 4x float4.
__device__ __forceinline__ v16bf load_b_f32_k1(const float* __restrict__ base,
                                               long sn, int k0, int n0, int lane) {
  const int n  = n0 + (lane & 15);
  const int kh = lane >> 4;
  const float4* p = (const float4*)(base + (long)n * sn + k0 + kh * 16);
  const float4 x0 = p[0], x1 = p[1], x2 = p[2], x3 = p[3];
  v16bf b;
  b[0]=(__bf16)x0.x; b[1]=(__bf16)x0.y; b[2]=(__bf16)x0.z; b[3]=(__bf16)x0.w;
  b[4]=(__bf16)x1.x; b[5]=(__bf16)x1.y; b[6]=(__bf16)x1.z; b[7]=(__bf16)x1.w;
  b[8]=(__bf16)x2.x; b[9]=(__bf16)x2.y; b[10]=(__bf16)x2.z; b[11]=(__bf16)x2.w;
  b[12]=(__bf16)x3.x; b[13]=(__bf16)x3.y; b[14]=(__bf16)x3.z; b[15]=(__bf16)x3.w;
  return b;
}

// B from f32 memory, strided k (element (k,n) = base[k*sk + n]): scalar loads.
__device__ __forceinline__ v16bf load_b_f32_strided(const float* __restrict__ base,
                                                    long sk, int k0, int n0, int lane) {
  const int n  = n0 + (lane & 15);
  const int kh = lane >> 4;
  const float* p = base + n + (long)(k0 + kh * 16) * sk;
  v16bf b;
#pragma unroll
  for (int v = 0; v < 8; ++v) {
    b[2 * v]     = (__bf16)p[(2 * v) * sk];
    b[2 * v + 1] = (__bf16)p[(2 * v + 1) * sk];
  }
  return b;
}

// ---------------------------------------------------------------------------
// TDM: DMA a [rows x 512] f32 2-D tile, global -> LDS.
// 6-arg builtin (clang-23 / therock-10.0 form):
//   (v4u group0, v8i group1, v4i group2, v4i group3, v8i extra, i32 cpol)
// Descriptor per CDNA5 ISA (groups 2/3 zero => 2-D tile).
// ---------------------------------------------------------------------------
__device__ __forceinline__ void tdm_load_rows_f32(unsigned int lds_off,
                                                  const float* gsrc, int rows) {
  const unsigned long long ga = (unsigned long long)(uintptr_t)gsrc;
  v4u g0;
  g0[0] = 1u;                                             // count=1, user D#
  g0[1] = lds_off;                                        // lds_addr (bytes)
  g0[2] = (unsigned int)(ga & 0xFFFFFFFFull);             // global_addr[31:0]
  g0[3] = (unsigned int)((ga >> 32) & 0x01FFFFFFull)      // global_addr[56:32]
        | (2u << 30);                                     // type = 2 ("image")
  v8i g1;
  g1[0] = (int)(2u << 16);                    // wg_mask=0, data_size=2 (4B)
  g1[1] = (int)((KR_ & 0xFFFF) << 16);        // tensor_dim0 = 512 (lo16)
  g1[2] = (int)((S_ & 0xFFFF) << 16);         // dim0 hi=0 | tensor_dim1 = 4096 (lo16)
  g1[3] = (int)((KR_ & 0xFFFF) << 16);        // dim1 hi=0 | tile_dim0 = 512
  g1[4] = rows;                               // tile_dim1 = rows, tile_dim2 = 0
  g1[5] = KR_;                                // tensor_dim0_stride = 512
  g1[6] = 0;
  g1[7] = 0;
  const v4i gz4 = {0, 0, 0, 0};
  const v8i gz8 = {0, 0, 0, 0, 0, 0, 0, 0};
  __builtin_amdgcn_tensor_load_to_lds(g0, g1, gz4, gz4, gz8, 0);
}

// ---------------------------------------------------------------------------
// Kernel 1: per head h: q_h = hidden @ w_q_h^T + b_q (32x128) -> LDS bf16,
//           q_abs_h = q_h @ w_kc_h (32x512) -> ws.  grid=128, 256 thr (8 waves)
// ---------------------------------------------------------------------------
__global__ void __launch_bounds__(256)
k_qproj_absorb(const float* __restrict__ hidden,  // [B, QR]
               const float* __restrict__ w_q,     // [H*D, QR]
               const float* __restrict__ b_q,     // [H*D]
               const float* __restrict__ w_kc,    // [H, D, KR]
               float* __restrict__ q_abs) {       // [B, H, KR]
  __shared__ __bf16 q_lds[B_ * D_];

  const int h    = blockIdx.x;
  const int lane = threadIdx.x & 31;
  const int wave = threadIdx.x >> 5;

  const float* wqh = w_q + (long)h * D_ * QR_;
#pragma unroll
  for (int mt = 0; mt < 2; ++mt) {
    v8f c = {};
    for (int k0 = 0; k0 < QR_; k0 += 32) {
      v16bf a  = load_a_f32v(hidden, QR_, mt * 16, k0, lane);
      v16bf bm = load_b_f32_k1(wqh, QR_, k0, wave * 16, lane);
      c = wmma_bf16(a, bm, c);
    }
    const int n     = wave * 16 + (lane & 15);
    const float bia = b_q[h * D_ + n];
#pragma unroll
    for (int v = 0; v < 8; ++v) {
      const int m = mt * 16 + (lane >> 4) * 8 + v;
      q_lds[m * D_ + n] = (__bf16)(c[v] + bia);
    }
  }
  __syncthreads();

  const float* wk = w_kc + (long)h * D_ * KR_;
#pragma unroll
  for (int t = 0; t < 8; ++t) {
    const int tile = wave * 8 + t;
    const int mt = tile & 1;
    const int nt = tile >> 1;
    v8f c = {};
#pragma unroll
    for (int k0 = 0; k0 < D_; k0 += 32) {
      v16bf a  = load_a_bf16v(q_lds, D_, mt * 16, k0, lane);
      v16bf bm = load_b_f32_strided(wk, KR_, k0, nt * 16, lane);  // w_kc[k*KR+n]
      c = wmma_bf16(a, bm, c);
    }
    const int n = nt * 16 + (lane & 15);
#pragma unroll
    for (int v = 0; v < 8; ++v) {
      const int m = mt * 16 + (lane >> 4) * 8 + v;
      q_abs[(long)m * (H_ * KR_) + (long)h * KR_ + n] = c[v];
    }
  }
}

// ---------------------------------------------------------------------------
// Kernel 2: fused flash attention; KV tile f32 in LDS via TDM.
// block = (b, head-group of 16), grid = 32*8 = 256, 256 threads.
// LDS: kv f32 [128][512] (256KB) + qa bf16 [16][512] (16KB) + probs bf16 (4KB).
// ---------------------------------------------------------------------------
#define HG_ 16
#define ST_ 128

#define KV_BYTES_  (ST_ * KR_ * 4)
#define QA_BYTES_  (HG_ * KR_ * 2)
#define PB_BYTES_  (HG_ * ST_ * 2)

__global__ void __launch_bounds__(256)
k_attention(const float* __restrict__ q_abs,  // [B, H, KR]
            const float* __restrict__ kv,     // [B, S, KR]
            float* __restrict__ ctx) {        // [B, H, KR]
  extern __shared__ char smem[];
  float*  kv_lds = (float*)smem;                                   // [ST][KR] f32
  __bf16* qa_lds = (__bf16*)(smem + KV_BYTES_);                    // [HG][KR]
  __bf16* p_lds  = (__bf16*)(smem + KV_BYTES_ + QA_BYTES_);        // [HG][ST]
  float*  wmax   = (float*)(smem + KV_BYTES_ + QA_BYTES_ + PB_BYTES_);  // [8][16]
  float*  wsum    = wmax + 128;
  float*  run_max = wsum + 128;
  float*  run_sum = run_max + 16;
  float*  scale_s = run_sum + 16;

  const int b    = blockIdx.x >> 3;
  const int hg   = blockIdx.x & 7;
  const int tid  = threadIdx.x;
  const int lane = tid & 31;
  const int wave = tid >> 5;

  const float* kvb = kv + (long)b * S_ * KR_;
  const unsigned int kv_off = (unsigned int)(uintptr_t)kv_lds;

  // Kick off DMA of the first KV tile (each wave owns a 16-row slice).
  tdm_load_rows_f32(kv_off + wave * 16 * KR_ * 4,
                    kvb + (long)wave * 16 * KR_, 16);

  // Stage q_absorbed as bf16 (packed 64-bit stores).
  const float4* qs = (const float4*)(q_abs + ((long)b * H_ + hg * HG_) * KR_);
  uint2* qd = (uint2*)qa_lds;
  for (int i = tid; i < HG_ * KR_ / 4; i += 256) {
    const float4 x = qs[i];
    BF4 t;
    t.h[0] = (__bf16)x.x; t.h[1] = (__bf16)x.y;
    t.h[2] = (__bf16)x.z; t.h[3] = (__bf16)x.w;
    qd[i] = t.u;
  }
  if (tid < HG_) { run_max[tid] = -3.0e38f; run_sum[tid] = 0.0f; }

  v8f acc[4] = {};
  const int hh = (lane >> 4) * 8;

  for (int s0 = 0; s0 < S_; s0 += ST_) {
    __builtin_amdgcn_s_wait_tensorcnt(0);   // this wave's slice has landed
    __syncthreads();                        // all slices have landed

    // --- Scores: M=16 heads, N = 16 cols at s0 + wave*16, K = 512.
    v8f sc = {};
    for (int k0 = 0; k0 < KR_; k0 += 32) {
      v16bf a  = load_a_bf16v(qa_lds, KR_, 0, k0, lane);
      v16bf bm = load_b_f32_k1(kv_lds, KR_, k0, wave * 16, lane);  // kv[s][r]
      sc = wmma_bf16(a, bm, sc);
    }

    // --- Per-head tile max (reduce 16 columns across the half-wave).
    {
      float pm[8];
#pragma unroll
      for (int v = 0; v < 8; ++v) {
        float m = sc[v];
        for (int off = 8; off >= 1; off >>= 1) m = fmaxf(m, __shfl_xor(m, off, 16));
        pm[v] = m;
      }
      if ((lane & 15) == 0) {
#pragma unroll
        for (int v = 0; v < 8; ++v) wmax[wave * 16 + hh + v] = pm[v];
      }
    }
    __syncthreads();

    if (tid < HG_) {
      const float old = run_max[tid];
      float tm = old;
      for (int w = 0; w < 8; ++w) tm = fmaxf(tm, wmax[w * 16 + tid]);
      const float corr = __expf(old - tm);
      run_max[tid] = tm;
      scale_s[tid] = corr;
      run_sum[tid] *= corr;
    }
    __syncthreads();

    // --- Exponentiate, write probs (bf16), partial sums.
    {
      const int n = lane & 15;
      float ps[8];
#pragma unroll
      for (int v = 0; v < 8; ++v) {
        const int hr = hh + v;
        const float p = __expf(sc[v] - run_max[hr]);
        p_lds[hr * ST_ + wave * 16 + n] = (__bf16)p;
        float s = p;
        for (int off = 8; off >= 1; off >>= 1) s += __shfl_xor(s, off, 16);
        ps[v] = s;
      }
      if (n == 0) {
#pragma unroll
        for (int v = 0; v < 8; ++v) wsum[wave * 16 + hh + v] = ps[v];
      }
    }
    __syncthreads();
    if (tid < HG_) {
      float s = run_sum[tid];
      for (int w = 0; w < 8; ++w) s += wsum[w * 16 + tid];
      run_sum[tid] = s;
    }

    // --- Context: rescale accumulators, acc += probs[16xST] @ kv[ST x 512].
    {
      float cs[8];
#pragma unroll
      for (int v = 0; v < 8; ++v) cs[v] = scale_s[hh + v];
#pragma unroll
      for (int i = 0; i < 4; ++i) {
#pragma unroll
        for (int v = 0; v < 8; ++v) acc[i][v] *= cs[v];
        const int n0 = (wave * 4 + i) * 16;
#pragma unroll
        for (int k0 = 0; k0 < ST_; k0 += 32) {
          v16bf a  = load_a_bf16v(p_lds, ST_, 0, k0, lane);
          v16bf bm = load_b_f32_strided(kv_lds, KR_, k0, n0, lane);  // kv[k*KR+n]
          acc[i] = wmma_bf16(a, bm, acc[i]);
        }
      }
    }
    __syncthreads();  // everyone done reading kv_lds / p_lds

    // DMA next KV tile while nothing depends on it yet.
    if (s0 + ST_ < S_) {
      tdm_load_rows_f32(kv_off + wave * 16 * KR_ * 4,
                        kvb + (long)(s0 + ST_ + wave * 16) * KR_, 16);
    }
  }

  // --- Finalize: context / sum -> ws.
  {
    const int n = lane & 15;
    float inv[8];
#pragma unroll
    for (int v = 0; v < 8; ++v) inv[v] = 1.0f / run_sum[hh + v];
    float* dst = ctx + ((long)b * H_ + hg * HG_) * KR_;
#pragma unroll
    for (int i = 0; i < 4; ++i) {
      const int n0 = (wave * 4 + i) * 16;
#pragma unroll
      for (int v = 0; v < 8; ++v) {
        dst[(long)(hh + v) * KR_ + n0 + n] = acc[i][v] * inv[v];
      }
    }
  }
}

// ---------------------------------------------------------------------------
// Kernel 3: out[b, h*128+d] = sum_r ctx[b,h,r] * w_vc[h,d,r]
// per head: M=32 (b), N=128 (d), K=512 (r). 2048 tiles -> 256 blocks x 8 waves.
// ---------------------------------------------------------------------------
__global__ void __launch_bounds__(256)
k_outproj(const float* __restrict__ ctx,   // [B, H, KR]
          const float* __restrict__ w_vc,  // [H, D, KR]
          float* __restrict__ out) {       // [B, H*D]
  const int h    = blockIdx.x >> 1;
  const int t    = (blockIdx.x & 1) * 8 + (threadIdx.x >> 5);
  const int lane = threadIdx.x & 31;
  const int mt = t & 1;
  const int nt = t >> 1;

  const float* wv = w_vc + (long)h * D_ * KR_;
  v8f c = {};
  for (int k0 = 0; k0 < KR_; k0 += 32) {
    v16bf a  = load_a_f32v(ctx + (long)h * KR_, (long)H_ * KR_, mt * 16, k0, lane);
    v16bf bm = load_b_f32_k1(wv, KR_, k0, nt * 16, lane);
    c = wmma_bf16(a, bm, c);
  }
  const int n = nt * 16 + (lane & 15);
#pragma unroll
  for (int v = 0; v < 8; ++v) {
    const int m = mt * 16 + (lane >> 4) * 8 + v;
    out[(long)m * (H_ * D_) + h * D_ + n] = c[v];
  }
}

// ---------------------------------------------------------------------------
extern "C" void kernel_launch(void* const* d_in, const int* in_sizes, int n_in,
                              void* d_out, int out_size, void* d_ws, size_t ws_size,
                              hipStream_t stream) {
  (void)in_sizes; (void)n_in; (void)out_size; (void)ws_size;
  const float* hidden = (const float*)d_in[0];  // [B, QR]
  const float* kv     = (const float*)d_in[1];  // [B, S, KR]
  const float* w_q    = (const float*)d_in[2];  // [H*D, QR]
  const float* b_q    = (const float*)d_in[3];  // [H*D]
  const float* w_kc   = (const float*)d_in[4];  // [H, D, KR]
  const float* w_vc   = (const float*)d_in[5];  // [H, D, KR]
  float* out = (float*)d_out;

  float* qa  = (float*)d_ws;                // 32*128*512 f32 = 8MB
  float* ctx = qa + (size_t)B_ * H_ * KR_;  // 8MB more

  k_qproj_absorb<<<H_, 256, 0, stream>>>(hidden, w_q, b_q, w_kc, qa);

  const size_t sh = KV_BYTES_ + QA_BYTES_ + PB_BYTES_
                  + sizeof(float) * (128 + 128 + 16 + 16 + 16);
  k_attention<<<B_ * (H_ / HG_), 256, sh, stream>>>(qa, kv, ctx);

  k_outproj<<<H_ * 2, 256, 0, stream>>>(ctx, w_vc, out);
}
